// ScaledSpatialGNN_17188459119261
// MI455X (gfx1250) — matrix operand order repaired and checked by reading, not compiled, not run on hardware
//
#include <hip/hip_runtime.h>
#include <hip/hip_bf16.h>

typedef __attribute__((ext_vector_type(16))) _Float16 v16h;
typedef __attribute__((ext_vector_type(8)))  _Float16 v8h;
typedef __attribute__((ext_vector_type(8)))  float    v8f;

union FragH { v16h v; v8h h[2]; };

#define EPS 1e-5f

// ---------------------------------------------------------------------------
// Degree / normalization
// ---------------------------------------------------------------------------
__global__ void deg_init(float* deg, int n) {
    int i = blockIdx.x * blockDim.x + threadIdx.x;
    if (i < n) deg[i] = 1.0f;               // self-loop
}

__global__ void deg_edges(const long long* __restrict__ ei, float* deg, int E) {
    int e = blockIdx.x * blockDim.x + threadIdx.x;
    if (e < E) atomicAdd(&deg[(int)ei[(size_t)E + e]], 1.0f);
}

__global__ void deg_rsqrt(float* deg, int n) {
    int i = blockIdx.x * blockDim.x + threadIdx.x;
    if (i < n) deg[i] = rsqrtf(deg[i]);
}

__global__ void zero_f32(float* p, int n) {
    int i = blockIdx.x * blockDim.x + threadIdx.x;
    if (i < n) p[i] = 0.0f;
}

// ---------------------------------------------------------------------------
// Input LayerNorm over D=512, emit f16 activations
// ---------------------------------------------------------------------------
__global__ __launch_bounds__(128) void ln_input(const float* __restrict__ x,
                                                const float* __restrict__ g,
                                                const float* __restrict__ b,
                                                _Float16* __restrict__ out, int N) {
    constexpr int D = 512;
    __shared__ float red[128];
    int row = blockIdx.x;
    if (row >= N) return;
    const float4* xr = (const float4*)(x + (size_t)row * D);
    int t = threadIdx.x;                     // 128 threads, one float4 each
    float4 v = xr[t];
    red[t] = v.x + v.y + v.z + v.w;
    __syncthreads();
    for (int off = 64; off; off >>= 1) { if (t < off) red[t] += red[t + off]; __syncthreads(); }
    float mu = red[0] * (1.0f / D);
    __syncthreads();
    float dx = v.x - mu, dy = v.y - mu, dz = v.z - mu, dw = v.w - mu;
    red[t] = dx * dx + dy * dy + dz * dz + dw * dw;
    __syncthreads();
    for (int off = 64; off; off >>= 1) { if (t < off) red[t] += red[t + off]; __syncthreads(); }
    float var = red[0] * (1.0f / D);
    float r = rsqrtf(var + EPS);
    int c = t * 4;
    _Float16* o = out + (size_t)row * D + c;
    o[0] = (_Float16)(dx * r * g[c + 0] + b[c + 0]);
    o[1] = (_Float16)(dy * r * g[c + 1] + b[c + 1]);
    o[2] = (_Float16)(dz * r * g[c + 2] + b[c + 2]);
    o[3] = (_Float16)(dw * r * g[c + 3] + b[c + 3]);
}

// ---------------------------------------------------------------------------
// Repack f32 weight [K,F] -> f16 fragment-native layout:
//   Bp[(ktile*F + col)*32 + k%32], ktile = k/32
// So one 16x16 B-fragment (cols j*16..j*16+15, 32 K values) is per-lane
// contiguous: lane n reads 8 halves at ((ktile*F + col)*32 + 8*half) and 8 more at +16.
// ---------------------------------------------------------------------------
__global__ void pack_w_f16(const float* __restrict__ W, _Float16* __restrict__ Bp,
                           int K, int F) {
    int t = blockIdx.x * blockDim.x + threadIdx.x;
    if (t >= K * F) return;
    int k = t / F, col = t % F;
    int ktile = k >> 5, kin = k & 31;
    Bp[((size_t)(ktile * F + col) << 5) + kin] = (_Float16)W[t];
}

// ---------------------------------------------------------------------------
// WMMA GEMM: C[M,F] = A[M,K] (f16, row major) @ Bpack (f16), f32 accumulate.
// Block = 256 threads = 8 waves. Block tile = 128 rows; wave w owns rows
// [blk*128 + 16w, +16) and ALL F columns (NF = F/16 fragments).
// A fragment layout (ISA 16-bit A 16x32): lane half h, row m = lane&15,
//   elements 0..7  -> K = kk + 8h + 0..7
//   elements 8..15 -> K = kk + 16 + 8h + 0..7     (two 16-B loads)
// ---------------------------------------------------------------------------
template <int K, int NF>
__global__ __launch_bounds__(256) void gemm_wmma(const _Float16* __restrict__ A,
                                                 const _Float16* __restrict__ Bp,
                                                 float* __restrict__ C, int M) {
    constexpr int F = NF * 16;
    const int wave = threadIdx.x >> 5;
    const int lane = threadIdx.x & 31;
    const int half = lane >> 4;
    const int ml   = lane & 15;
    const int rowBase = blockIdx.x * 128 + wave * 16;
    int m = rowBase + ml;
    int mc = m < M ? m : (M - 1);

    v8f zero = {0.f, 0.f, 0.f, 0.f, 0.f, 0.f, 0.f, 0.f};
    v8f acc[NF];
#pragma unroll
    for (int j = 0; j < NF; ++j) acc[j] = zero;

    const _Float16* arow = A + (size_t)mc * K;

    for (int kk = 0; kk < K; kk += 32) {
        FragH af;
        af.h[0] = *(const v8h*)(arow + kk + 8 * half);
        af.h[1] = *(const v8h*)(arow + kk + 16 + 8 * half);

        const _Float16* bbase = Bp + (((size_t)(kk >> 5) * F) << 5);
#pragma unroll
        for (int j = 0; j < NF; ++j) {
            FragH bf;
            const _Float16* p = bbase + ((size_t)(j * 16 + ml) << 5) + 8 * half;
            bf.h[0] = *(const v8h*)(p);
            bf.h[1] = *(const v8h*)(p + 16);
            acc[j] = __builtin_amdgcn_wmma_f32_16x16x32_f16(
                false, af.v, false, bf.v, (short)0, acc[j], false, false);
        }
    }

    // C/D layout: element r -> row = rowBase + r + 8*half, col = j*16 + (lane&15)
#pragma unroll
    for (int j = 0; j < NF; ++j) {
#pragma unroll
        for (int r = 0; r < 8; ++r) {
            int mo = rowBase + r + 8 * half;
            if (mo < M) C[(size_t)mo * F + j * 16 + ml] = acc[j][r];
        }
    }
}

// ---------------------------------------------------------------------------
// GCN aggregation: agg = lin*dis^2 + bias  (self loop), then edge atomics
// ---------------------------------------------------------------------------
template <int F>
__global__ void selfloop_init(const float* __restrict__ lin, const float* __restrict__ dis,
                              const float* __restrict__ bias, float* __restrict__ agg, int N) {
    int t = blockIdx.x * blockDim.x + threadIdx.x;
    int v = t / F, f = t % F;
    if (v >= N) return;
    float d = dis[v];
    agg[t] = lin[t] * d * d + bias[f];
}

template <int F>
__global__ void edge_scatter(const long long* __restrict__ ei,
                             const float* __restrict__ lin,
                             const float* __restrict__ dis,
                             float* __restrict__ agg, int E) {
    int t = blockIdx.x * blockDim.x + threadIdx.x;
    int e = t / F, f = t % F;
    if (e >= E) return;
    int src = (int)ei[e];
    int dst = (int)ei[(size_t)E + e];
    float w = dis[src] * dis[dst];
    atomicAdd(&agg[(size_t)dst * F + f], lin[(size_t)src * F + f] * w);
}

// ---------------------------------------------------------------------------
// BatchNorm over nodes: per-feature sum / sumsq, then normalize + ReLU -> f16
// ---------------------------------------------------------------------------
template <int F>
__global__ void bn_stats(const float* __restrict__ agg, float* __restrict__ sums,
                         float* __restrict__ sqs, int N) {
    int f = threadIdx.x;                    // blockDim.x == F
    float s = 0.f, q = 0.f;
    for (int v = blockIdx.x; v < N; v += gridDim.x) {
        float x = agg[(size_t)v * F + f];
        s += x; q += x * x;
    }
    atomicAdd(&sums[f], s);
    atomicAdd(&sqs[f], q);
}

template <int F>
__global__ void bn_relu_f16(const float* __restrict__ agg, const float* __restrict__ sums,
                            const float* __restrict__ sqs, const float* __restrict__ g,
                            const float* __restrict__ b, _Float16* __restrict__ out, int N) {
    int t = blockIdx.x * blockDim.x + threadIdx.x;
    int v = t / F, f = t % F;
    if (v >= N) return;
    float invN = 1.0f / (float)N;
    float mu  = sums[f] * invN;
    float var = sqs[f] * invN - mu * mu;
    float y = (agg[t] - mu) * rsqrtf(var + EPS) * g[f] + b[f];
    out[t] = (_Float16)fmaxf(y, 0.0f);
}

// ---------------------------------------------------------------------------
// Fused MLP head: 32 -> 16 (LN,ReLU) -> 8 (LN,ReLU) -> 8, one thread per node
// ---------------------------------------------------------------------------
__global__ __launch_bounds__(256) void mlp_head(const _Float16* __restrict__ act,
        const float* __restrict__ Wc1, const float* __restrict__ bc1,
        const float* __restrict__ g1,  const float* __restrict__ b1,
        const float* __restrict__ Wc2, const float* __restrict__ bc2,
        const float* __restrict__ g2,  const float* __restrict__ b2,
        const float* __restrict__ Wc3, const float* __restrict__ bc3,
        float* __restrict__ out, int N) {
    int v = blockIdx.x * blockDim.x + threadIdx.x;
    if (v >= N) return;
    const _Float16* a = act + (size_t)v * 32;
    float h0[32], h1[16], h2[8];
#pragma unroll
    for (int i = 0; i < 32; ++i) h0[i] = (float)a[i];

    // 32 -> 16, LN(16), ReLU
#pragma unroll
    for (int j = 0; j < 16; ++j) {
        float s = bc1[j];
#pragma unroll
        for (int i = 0; i < 32; ++i) s += h0[i] * Wc1[i * 16 + j];
        h1[j] = s;
    }
    float mu = 0.f;
#pragma unroll
    for (int j = 0; j < 16; ++j) mu += h1[j];
    mu *= (1.0f / 16.0f);
    float var = 0.f;
#pragma unroll
    for (int j = 0; j < 16; ++j) { float d = h1[j] - mu; var += d * d; }
    var *= (1.0f / 16.0f);
    float r = rsqrtf(var + EPS);
#pragma unroll
    for (int j = 0; j < 16; ++j) h1[j] = fmaxf((h1[j] - mu) * r * g1[j] + b1[j], 0.0f);

    // 16 -> 8, LN(8), ReLU
#pragma unroll
    for (int j = 0; j < 8; ++j) {
        float s = bc2[j];
#pragma unroll
        for (int i = 0; i < 16; ++i) s += h1[i] * Wc2[i * 8 + j];
        h2[j] = s;
    }
    mu = 0.f;
#pragma unroll
    for (int j = 0; j < 8; ++j) mu += h2[j];
    mu *= (1.0f / 8.0f);
    var = 0.f;
#pragma unroll
    for (int j = 0; j < 8; ++j) { float d = h2[j] - mu; var += d * d; }
    var *= (1.0f / 8.0f);
    r = rsqrtf(var + EPS);
#pragma unroll
    for (int j = 0; j < 8; ++j) h2[j] = fmaxf((h2[j] - mu) * r * g2[j] + b2[j], 0.0f);

    // 8 -> 8
    float* o = out + (size_t)v * 8;
#pragma unroll
    for (int j = 0; j < 8; ++j) {
        float s = bc3[j];
#pragma unroll
        for (int i = 0; i < 8; ++i) s += h2[i] * Wc3[i * 8 + j];
        o[j] = s;
    }
}

// ---------------------------------------------------------------------------
// Host launcher
// ---------------------------------------------------------------------------
extern "C" void kernel_launch(void* const* d_in, const int* in_sizes, int n_in,
                              void* d_out, int out_size, void* d_ws, size_t ws_size,
                              hipStream_t stream) {
    const float*     x    = (const float*)d_in[0];
    const long long* ei   = (const long long*)d_in[1];
    const float *ln_g = (const float*)d_in[2],  *ln_b = (const float*)d_in[3];
    const float *W1 = (const float*)d_in[4],  *b1 = (const float*)d_in[5];
    const float *bn1g = (const float*)d_in[6], *bn1b = (const float*)d_in[7];
    const float *W2 = (const float*)d_in[8],  *b2 = (const float*)d_in[9];
    const float *bn2g = (const float*)d_in[10], *bn2b = (const float*)d_in[11];
    const float *W3 = (const float*)d_in[12], *b3 = (const float*)d_in[13];
    const float *bn3g = (const float*)d_in[14], *bn3b = (const float*)d_in[15];
    const float *W4 = (const float*)d_in[16], *b4 = (const float*)d_in[17];
    const float *bn4g = (const float*)d_in[18], *bn4b = (const float*)d_in[19];
    const float *Wc1 = (const float*)d_in[20], *bc1 = (const float*)d_in[21];
    const float *g1  = (const float*)d_in[22], *bb1 = (const float*)d_in[23];
    const float *Wc2 = (const float*)d_in[24], *bc2 = (const float*)d_in[25];
    const float *g2  = (const float*)d_in[26], *bb2 = (const float*)d_in[27];
    const float *Wc3 = (const float*)d_in[28], *bc3 = (const float*)d_in[29];
    float* out = (float*)d_out;

    const int N = in_sizes[0] / 512;
    const int E = in_sizes[1] / 2;

    // workspace layout
    char* ws = (char*)d_ws;
    size_t off = 0;
    auto take = [&](size_t bytes) -> char* {
        char* p = ws + off;
        off = (off + bytes + 255) & ~(size_t)255;
        return p;
    };
    float*     dis   = (float*)take((size_t)N * 4);            // deg -> dis in place
    _Float16*  xa    = (_Float16*)take((size_t)N * 512 * 2);   // activations (f16), reused
    float*     lin   = (float*)take((size_t)N * 128 * 4);      // GEMM output
    float*     agg   = (float*)take((size_t)N * 128 * 4);      // aggregated
    float*     stats = (float*)take(256 * 4);                  // sums | sqs
    _Float16*  W1p   = (_Float16*)take((size_t)512 * 128 * 2);
    _Float16*  W2p   = (_Float16*)take((size_t)128 * 128 * 2);
    _Float16*  W3p   = (_Float16*)take((size_t)128 * 64 * 2);
    _Float16*  W4p   = (_Float16*)take((size_t)64 * 32 * 2);
    float* sums = stats;
    float* sqs  = stats + 128;

    const int T = 256;
    auto cdiv = [](long long a, long long b) { return (int)((a + b - 1) / b); };

    // degree normalization
    deg_init<<<cdiv(N, T), T, 0, stream>>>(dis, N);
    deg_edges<<<cdiv(E, T), T, 0, stream>>>(ei, dis, E);
    deg_rsqrt<<<cdiv(N, T), T, 0, stream>>>(dis, N);

    // input layer norm -> f16
    ln_input<<<N, 128, 0, stream>>>(x, ln_g, ln_b, xa, N);

    // weight repack
    pack_w_f16<<<cdiv(512 * 128, T), T, 0, stream>>>(W1, W1p, 512, 128);
    pack_w_f16<<<cdiv(128 * 128, T), T, 0, stream>>>(W2, W2p, 128, 128);
    pack_w_f16<<<cdiv(128 * 64,  T), T, 0, stream>>>(W3, W3p, 128, 64);
    pack_w_f16<<<cdiv(64 * 32,   T), T, 0, stream>>>(W4, W4p, 64, 32);

    const int gemmGrid = cdiv(N, 128);

#define GCN_LAYER(KDIM, FDIM, WP, BIAS, BNG, BNB)                                        \
    do {                                                                                  \
        gemm_wmma<KDIM, (FDIM) / 16><<<gemmGrid, 256, 0, stream>>>(xa, WP, lin, N);       \
        selfloop_init<FDIM><<<cdiv((long long)N * (FDIM), T), T, 0, stream>>>(            \
            lin, dis, BIAS, agg, N);                                                      \
        edge_scatter<FDIM><<<cdiv((long long)E * (FDIM), T), T, 0, stream>>>(             \
            ei, lin, dis, agg, E);                                                        \
        zero_f32<<<1, 256, 0, stream>>>(stats, 256);                                      \
        bn_stats<FDIM><<<512, FDIM, 0, stream>>>(agg, sums, sqs, N);                      \
        bn_relu_f16<FDIM><<<cdiv((long long)N * (FDIM), T), T, 0, stream>>>(              \
            agg, sums, sqs, BNG, BNB, xa, N);                                             \
    } while (0)

    GCN_LAYER(512, 128, W1p, b1, bn1g, bn1b);
    GCN_LAYER(128, 128, W2p, b2, bn2g, bn2b);
    GCN_LAYER(128,  64, W3p, b3, bn3g, bn3b);
    GCN_LAYER(64,   32, W4p, b4, bn4g, bn4b);
#undef GCN_LAYER

    // fused MLP head -> out [N, 8]
    mlp_head<<<cdiv(N, T), T, 0, stream>>>(xa, Wc1, bc1, g1, bb1,
                                           Wc2, bc2, g2, bb2, Wc3, bc3, out, N);
}
